// MultiheadDifferenceGeneralAttention_39341900431914
// MI455X (gfx1250) — matrix operand order repaired and checked by reading, not compiled, not run on hardware
//
#include <hip/hip_runtime.h>

// ---------------------------------------------------------------------------
// Agent-aware multihead attention for MI455X (gfx1250), bf16 WMMA pipeline.
// L=1024, N=4, E=1024, H=16, hd=64. All matmuls use v_wmma_f32_16x16x32_bf16.
// GEMM A-tiles are staged into LDS with the Tensor Data Mover
// (tensor_load_to_lds + s_wait_tensorcnt); in-projection scatters into
// per-head compact layouts so every attention WMMA operand is a contiguous
// 32B-per-lane load.
// ---------------------------------------------------------------------------

typedef __attribute__((ext_vector_type(16))) __bf16 v16bf;
typedef __attribute__((ext_vector_type(8)))  __bf16 v8bf;
typedef __attribute__((ext_vector_type(8)))  float  v8f;
typedef __attribute__((ext_vector_type(4)))  unsigned int v4u;
typedef __attribute__((ext_vector_type(8)))  int v8i;
typedef __attribute__((ext_vector_type(4)))  int v4i;

#define LSEQ   1024
#define NB     4
#define EMB    1024
#define NHEAD  16
#define HD     64
#define F5     5120     // 5*E
#define MROWS  4096     // L*N

__device__ __forceinline__ v8f wmma_bf16(v16bf a, v16bf b, v8f c) {
  return __builtin_amdgcn_wmma_f32_16x16x32_bf16(false, a, false, b, (short)0, c,
                                                 false, false);
}

// A-matrix fragment (16x32 bf16), contiguous K order matching load_b_frag:
//   lane&15 = M row; halves h -> K = k0 + 16*(lane>=16) + h.
__device__ __forceinline__ v16bf load_a_frag(const __bf16* base, size_t row_stride,
                                             int lane, int k0) {
  const int lo = lane & 15;
  const int c0 = (lane >> 4) << 4;           // 0 or 16
  const __bf16* p = base + (size_t)lo * row_stride + (size_t)(k0 + c0);
  return *(const v16bf*)p;
}

// B-matrix fragment (32x16 bf16), N-major memory (row index = N, K contiguous):
//   lane&15 = N; halves h -> K = k0 + 16*(lane>=16) + h  (32B contiguous load).
__device__ __forceinline__ v16bf load_b_frag(const __bf16* base, size_t row_stride,
                                             int lane, int k0) {
  const __bf16* p = base + (size_t)(lane & 15) * row_stride +
                    (size_t)(k0 + ((lane >> 4) << 4));
  return *(const v16bf*)p;
}

// ---------------------------------------------------------------------------
// Tensor Data Mover: DMA a 32x1024 bf16 tile (rows m0..m0+31 of a 4096x1024
// row-major matrix) into LDS.  D# per CDNA5 ISA 8.3-8.5:
//   group0: count=1 | lds_addr | global_addr[56:0] | type=2
//   group1: data_size=1(2B), tensor_dim0=1024, tensor_dim1=4096,
//           tile_dim0=1024, tile_dim1=32, tensor_dim0_stride=1024
//   groups 2/3: zero (2-D tile)
// Issued by one wave; completion via s_wait_tensorcnt, visibility via barrier.
// ---------------------------------------------------------------------------
__device__ __forceinline__ void tdm_load_a_tile(const __bf16* A, int m0,
                                                __bf16* lds_tile) {
  unsigned long long ga =
      (unsigned long long)(uintptr_t)(A + (size_t)m0 * 1024);
  unsigned int lds = (unsigned int)(uintptr_t)lds_tile;  // low 32b = LDS offset
  v4u g0;
  g0[0] = 1u;                                   // count=1 (valid descriptor)
  g0[1] = lds;                                  // lds_addr
  g0[2] = (unsigned int)ga;                     // global_addr[31:0]
  g0[3] = (unsigned int)((ga >> 32) & 0x01ffffffu) | (2u << 30);  // type=2
  v8i g1;
  g1[0] = 0x00010000;                           // data_size=1 -> 2 bytes
  g1[1] = (int)(1024u << 16);                   // tensor_dim0[15:0] at [63:48]
  g1[2] = (int)(4096u << 16);                   // tensor_dim1[15:0] at [95:80]
  g1[3] = (int)(1024u << 16);                   // tile_dim0 at [127:112]
  g1[4] = 32;                                   // tile_dim1 = 32 rows
  g1[5] = 1024;                                 // tensor_dim0_stride[31:0]
  g1[6] = 0;
  g1[7] = 0;
  v4i z4 = {0, 0, 0, 0};
#if __clang_major__ >= 23
  v8i z8 = {0, 0, 0, 0, 0, 0, 0, 0};
  __builtin_amdgcn_tensor_load_to_lds(g0, g1, z4, z4, z8, 0);
#else
  __builtin_amdgcn_tensor_load_to_lds(g0, g1, z4, z4, 0);
#endif
}

// ---------------------------------------------------------------------------
// fp32 -> bf16 conversion
// ---------------------------------------------------------------------------
__global__ void cvt_bf16_kernel(const float* __restrict__ src,
                                __bf16* __restrict__ dst, int n) {
  int i = blockIdx.x * blockDim.x + threadIdx.x;
  if (i < n) dst[i] = (__bf16)src[i];
}

// ---------------------------------------------------------------------------
// In-projection GEMM: proj = A(4096x1024) x Wi^T(5120x1024) + bias, with the
// epilogue scattering into per-head compact layouts:
//   Qc/Qsc/Kc/Ksc[(n*16+h)*1024 + l][d]   (row stride 64, d contiguous)
//   Vt [(n*16+h)*64 + d][s=l]             (row stride 1024, s contiguous)
// q / q_same columns scaled by hd^-0.5 = 0.125.
// Block: 256 thr = 8 waves; block tile 32x256; each wave owns a 2x2 tile grid
// (4 independent WMMA accumulator chains).  A tile staged in LDS by the TDM.
// ---------------------------------------------------------------------------
__global__ __launch_bounds__(256) void gemm_inproj_kernel(
    const __bf16* __restrict__ A, const __bf16* __restrict__ B,
    const float* __restrict__ bias,
    __bf16* __restrict__ Qc, __bf16* __restrict__ Qsc,
    __bf16* __restrict__ Kc, __bf16* __restrict__ Ksc,
    __bf16* __restrict__ Vt) {
  __shared__ __bf16 atile[32 * 1024];   // 64 KB

  const int tid  = threadIdx.x;
  const int wave = tid >> 5;
  const int lane = tid & 31;
  const int m0 = blockIdx.y << 5;
  const int n0 = (blockIdx.x << 8) + (wave << 5);

  if (wave == 0) {
    tdm_load_a_tile(A, m0, atile);
    __builtin_amdgcn_s_wait_tensorcnt(0);
  }
  __syncthreads();

  const __bf16* pb0 = B + (size_t)(n0 + (lane & 15)) * 1024 + ((lane >> 4) << 4);
  const __bf16* pb1 = pb0 + (size_t)16 * 1024;
  v8f acc00 = {}, acc01 = {}, acc10 = {}, acc11 = {};
#pragma unroll 2
  for (int k0 = 0; k0 < 1024; k0 += 32) {
    v16bf a0 = load_a_frag(atile, 1024, lane, k0);
    v16bf a1 = load_a_frag(atile + 16 * 1024, 1024, lane, k0);
    v16bf b0 = *(const v16bf*)(pb0 + k0);
    v16bf b1 = *(const v16bf*)(pb1 + k0);
    acc00 = wmma_bf16(a0, b0, acc00);
    acc01 = wmma_bf16(a0, b1, acc01);
    acc10 = wmma_bf16(a1, b0, acc10);
    acc11 = wmma_bf16(a1, b1, acc11);
    __builtin_prefetch(pb0 + k0 + 256, 0, 1);
    __builtin_prefetch(pb1 + k0 + 256, 0, 1);
  }

  const int mrow_off = (lane >> 4) << 3;
  const int ncol = lane & 15;
#pragma unroll
  for (int i = 0; i < 2; ++i) {
#pragma unroll
    for (int j = 0; j < 2; ++j) {
      v8f acc = (i == 0) ? ((j == 0) ? acc00 : acc01)
                         : ((j == 0) ? acc10 : acc11);
#pragma unroll
      for (int r = 0; r < 8; ++r) {
        int m = m0 + (i << 4) + r + mrow_off;
        int f = n0 + (j << 4) + ncol;
        float v = acc[r];
        if (f < 2048) v *= 0.125f;
        v += bias[f];
        int l = m >> 2, nb = m & 3;
        int stream = f >> 10;
        int w = f & 1023;
        int h = w >> 6, d = w & 63;
        __bf16 bv = (__bf16)v;
        size_t cidx = ((size_t)(nb * 16 + h) * 1024 + l) * 64 + d;
        if      (stream == 0) Qc [cidx] = bv;
        else if (stream == 1) Qsc[cidx] = bv;
        else if (stream == 2) Kc [cidx] = bv;
        else if (stream == 3) Ksc[cidx] = bv;
        else Vt[((size_t)(nb * 16 + h) * 64 + d) * 1024 + l] = bv;
      }
    }
  }
}

// ---------------------------------------------------------------------------
// Out-projection GEMM: out(4096x1024 fp32) = A(4096x1024 bf16) x Wo^T + bias.
// Same 32x256 block / 2x2-per-wave tiling; A tile staged by the TDM.
// ---------------------------------------------------------------------------
__global__ __launch_bounds__(256) void gemm_outproj_kernel(
    const __bf16* __restrict__ A, const __bf16* __restrict__ B,
    const float* __restrict__ bias, float* __restrict__ out) {
  __shared__ __bf16 atile[32 * 1024];

  const int tid  = threadIdx.x;
  const int wave = tid >> 5;
  const int lane = tid & 31;
  const int m0 = blockIdx.y << 5;
  const int n0 = (blockIdx.x << 8) + (wave << 5);

  if (wave == 0) {
    tdm_load_a_tile(A, m0, atile);
    __builtin_amdgcn_s_wait_tensorcnt(0);
  }
  __syncthreads();

  const __bf16* pb0 = B + (size_t)(n0 + (lane & 15)) * 1024 + ((lane >> 4) << 4);
  const __bf16* pb1 = pb0 + (size_t)16 * 1024;
  v8f acc00 = {}, acc01 = {}, acc10 = {}, acc11 = {};
#pragma unroll 2
  for (int k0 = 0; k0 < 1024; k0 += 32) {
    v16bf a0 = load_a_frag(atile, 1024, lane, k0);
    v16bf a1 = load_a_frag(atile + 16 * 1024, 1024, lane, k0);
    v16bf b0 = *(const v16bf*)(pb0 + k0);
    v16bf b1 = *(const v16bf*)(pb1 + k0);
    acc00 = wmma_bf16(a0, b0, acc00);
    acc01 = wmma_bf16(a0, b1, acc01);
    acc10 = wmma_bf16(a1, b0, acc10);
    acc11 = wmma_bf16(a1, b1, acc11);
    __builtin_prefetch(pb0 + k0 + 256, 0, 1);
    __builtin_prefetch(pb1 + k0 + 256, 0, 1);
  }

  const int mrow_off = (lane >> 4) << 3;
  const int ncol = lane & 15;
#pragma unroll
  for (int i = 0; i < 2; ++i) {
#pragma unroll
    for (int j = 0; j < 2; ++j) {
      v8f acc = (i == 0) ? ((j == 0) ? acc00 : acc01)
                         : ((j == 0) ? acc10 : acc11);
#pragma unroll
      for (int r = 0; r < 8; ++r) {
        int m = m0 + (i << 4) + r + mrow_off;
        int f = n0 + (j << 4) + ncol;
        out[(size_t)m * 1024 + f] = acc[r] + bias[f];
      }
    }
  }
}

// ---------------------------------------------------------------------------
// Fused agent-aware attention.
// Grid: (N, L/16). Block: 256 threads = 8 waves; wave w owns score columns
// [w*128, w*128+128). Loops over all 16 heads so the block exclusively owns
// its (n, l-tile, :) slice of weights_avg (no global atomics).
// ---------------------------------------------------------------------------
__global__ __launch_bounds__(256) void attn_kernel(
    const __bf16* __restrict__ Qc, const __bf16* __restrict__ Qsc,
    const __bf16* __restrict__ Kc, const __bf16* __restrict__ Ksc,
    const __bf16* __restrict__ Vt, const int* __restrict__ amask,
    float* __restrict__ wavg, __bf16* __restrict__ attnH) {
  __shared__ __bf16 wbuf[8][16][128];   // per-wave normalized weights (bf16)
  __shared__ float  red[16][8];         // cross-wave row reductions
  __shared__ float  oacc[16][64];       // per-head output accumulator

  const int tid  = threadIdx.x;
  const int wave = tid >> 5;
  const int lane = tid & 31;
  const int n  = blockIdx.x;
  const int l0 = blockIdx.y << 4;
  const int mrow_off = (lane >> 4) << 3;
  const int ncol = lane & 15;

  for (int h = 0; h < NHEAD; ++h) {
    for (int i = tid; i < 16 * 64; i += 256) ((float*)oacc)[i] = 0.f;
    __syncthreads();

    const size_t HB = (size_t)(n * 16 + h);
    const __bf16* qcb  = Qc  + (HB * 1024 + l0) * 64;
    const __bf16* qscb = Qsc + (HB * 1024 + l0) * 64;
    const __bf16* kcb  = Kc  + HB * 1024 * 64;
    const __bf16* kscb = Ksc + HB * 1024 * 64;
    const __bf16* vtb  = Vt  + HB * 64 * 1024;

    // q / q_same A fragments (K = 64 -> two x32 steps each)
    v16bf qa0 = load_a_frag(qcb,  64, lane, 0);
    v16bf qa1 = load_a_frag(qcb,  64, lane, 32);
    v16bf qs0 = load_a_frag(qscb, 64, lane, 0);
    v16bf qs1 = load_a_frag(qscb, 64, lane, 32);

    float sc[8][8];
    float pmax[8];
#pragma unroll
    for (int r = 0; r < 8; ++r) pmax[r] = -3.0e38f;

    // ---- dual-stream scores + mask combine -------------------------------
#pragma unroll
    for (int st = 0; st < 8; ++st) {
      const int s0 = (wave << 7) + (st << 4);
      const __bf16* kb  = kcb  + (size_t)s0 * 64;
      const __bf16* ksb = kscb + (size_t)s0 * 64;
      v8f aA = {}, aS = {};
      aA = wmma_bf16(qa0, load_b_frag(kb,  64, lane, 0),  aA);
      aA = wmma_bf16(qa1, load_b_frag(kb,  64, lane, 32), aA);
      aS = wmma_bf16(qs0, load_b_frag(ksb, 64, lane, 0),  aS);
      aS = wmma_bf16(qs1, load_b_frag(ksb, 64, lane, 32), aS);
#pragma unroll
      for (int r = 0; r < 8; ++r) {
        const int ml = r + mrow_off;
        float mk = (float)amask[(size_t)(l0 + ml) * LSEQ + (s0 + ncol)];
        float v  = mk * aS[r] + (1.f - mk) * aA[r];
        sc[st][r] = v;
        pmax[r] = fmaxf(pmax[r], v);
      }
    }

    // ---- softmax: row max -------------------------------------------------
#pragma unroll
    for (int r = 0; r < 8; ++r) {
      float v = pmax[r];
      v = fmaxf(v, __shfl_xor(v, 1, 32));
      v = fmaxf(v, __shfl_xor(v, 2, 32));
      v = fmaxf(v, __shfl_xor(v, 4, 32));
      v = fmaxf(v, __shfl_xor(v, 8, 32));
      pmax[r] = v;
    }
    if (ncol == 0) {
#pragma unroll
      for (int r = 0; r < 8; ++r) red[r + mrow_off][wave] = pmax[r];
    }
    __syncthreads();
    float rmax[8];
#pragma unroll
    for (int r = 0; r < 8; ++r) {
      const int ml = r + mrow_off;
      float v = red[ml][0];
#pragma unroll
      for (int w = 1; w < 8; ++w) v = fmaxf(v, red[ml][w]);
      rmax[r] = v;
    }
    __syncthreads();

    // ---- exp + row sum ----------------------------------------------------
    float psum[8];
#pragma unroll
    for (int r = 0; r < 8; ++r) psum[r] = 0.f;
#pragma unroll
    for (int st = 0; st < 8; ++st) {
#pragma unroll
      for (int r = 0; r < 8; ++r) {
        float e = __expf(sc[st][r] - rmax[r]);
        sc[st][r] = e;
        psum[r] += e;
      }
    }
#pragma unroll
    for (int r = 0; r < 8; ++r) {
      float v = psum[r];
      v += __shfl_xor(v, 1, 32);
      v += __shfl_xor(v, 2, 32);
      v += __shfl_xor(v, 4, 32);
      v += __shfl_xor(v, 8, 32);
      psum[r] = v;
    }
    if (ncol == 0) {
#pragma unroll
      for (int r = 0; r < 8; ++r) red[r + mrow_off][wave] = psum[r];
    }
    __syncthreads();
    float rsum[8];
#pragma unroll
    for (int r = 0; r < 8; ++r) {
      const int ml = r + mrow_off;
      float v = 0.f;
#pragma unroll
      for (int w = 0; w < 8; ++w) v += red[ml][w];
      rsum[r] = 1.f / v;
    }

    // ---- normalize, accumulate weights_avg, stash bf16 weights ------------
#pragma unroll
    for (int st = 0; st < 8; ++st) {
      const int s0 = (wave << 7) + (st << 4);
#pragma unroll
      for (int r = 0; r < 8; ++r) {
        const int ml = r + mrow_off;
        float w = sc[st][r] * rsum[r];
        size_t wi = (size_t)n * LSEQ * LSEQ + (size_t)(l0 + ml) * LSEQ + (s0 + ncol);
        float prev = (h == 0) ? 0.f : wavg[wi];
        wavg[wi] = prev + w * (1.f / (float)NHEAD);
        wbuf[wave][ml][(st << 4) + ncol] = (__bf16)w;
      }
    }

    // ---- PV: out(16x64) += weights(16x128) x V(128x64), per-wave K slice --
#pragma unroll
    for (int dt = 0; dt < 4; ++dt) {
      v8f po = {};
      const __bf16* vb = vtb + (size_t)(dt << 4) * 1024;   // rows = d, cols = s
#pragma unroll
      for (int kk = 0; kk < 4; ++kk) {
        v16bf a = load_a_frag(&wbuf[wave][0][0], 128, lane, kk << 5);
        v16bf b = load_b_frag(vb, 1024, lane, (wave << 7) + (kk << 5));
        po = wmma_bf16(a, b, po);
      }
#pragma unroll
      for (int r = 0; r < 8; ++r)
        atomicAdd(&oacc[r + mrow_off][(dt << 4) + ncol], po[r]);
    }
    __syncthreads();

    // ---- store per-head output (bf16) for out-projection ------------------
    for (int i = tid; i < 16 * 64; i += 256) {
      const int row = i >> 6, d = i & 63;
      attnH[(size_t)((l0 + row) * 4 + n) * EMB + h * HD + d] = (__bf16)oacc[row][d];
    }
    __syncthreads();
  }
}

// ---------------------------------------------------------------------------
// launch
// ---------------------------------------------------------------------------
extern "C" void kernel_launch(void* const* d_in, const int* in_sizes, int n_in,
                              void* d_out, int out_size, void* d_ws, size_t ws_size,
                              hipStream_t stream) {
  (void)in_sizes; (void)n_in; (void)out_size; (void)ws_size;
  const float* query = (const float*)d_in[0];
  const int*   amask = (const int*)d_in[1];
  const float* Wi    = (const float*)d_in[2];
  const float* bi    = (const float*)d_in[3];
  const float* Wo    = (const float*)d_in[4];
  const float* bo    = (const float*)d_in[5];

  char* ws = (char*)d_ws;
  __bf16* qbf   = (__bf16*)(ws);                          //  8 MB: 4096x1024
  __bf16* wibf  = (__bf16*)(ws + (8ull  << 20));          // 10 MB: 5120x1024
  __bf16* wobf  = (__bf16*)(ws + (18ull << 20));          //  2 MB: 1024x1024
  __bf16* Qc    = (__bf16*)(ws + (20ull << 20));          //  8 MB
  __bf16* Qsc   = (__bf16*)(ws + (28ull << 20));          //  8 MB
  __bf16* Kc    = (__bf16*)(ws + (36ull << 20));          //  8 MB
  __bf16* Ksc   = (__bf16*)(ws + (44ull << 20));          //  8 MB
  __bf16* Vt    = (__bf16*)(ws + (52ull << 20));          //  8 MB
  __bf16* attnH = (__bf16*)(ws + (60ull << 20));          //  8 MB

  float* outAttn = (float*)d_out;                          // (L,N,E) fp32
  float* wavg    = (float*)d_out + (size_t)MROWS * EMB;    // (N,L,L) fp32

  // fp32 -> bf16 conversions
  {
    int n1 = MROWS * EMB;
    cvt_bf16_kernel<<<(n1 + 255) / 256, 256, 0, stream>>>(query, qbf, n1);
    int n2 = F5 * EMB;
    cvt_bf16_kernel<<<(n2 + 255) / 256, 256, 0, stream>>>(Wi, wibf, n2);
    int n3 = EMB * EMB;
    cvt_bf16_kernel<<<(n3 + 255) / 256, 256, 0, stream>>>(Wo, wobf, n3);
  }

  // in-projection + scatter into per-head compact layouts
  gemm_inproj_kernel<<<dim3(F5 / 256, MROWS / 32), 256, 0, stream>>>(
      qbf, wibf, bi, Qc, Qsc, Kc, Ksc, Vt);

  // fused agent-aware attention (scores, softmax, weights_avg, PV)
  attn_kernel<<<dim3(NB, LSEQ / 16), 256, 0, stream>>>(
      Qc, Qsc, Kc, Ksc, Vt, amask, wavg, attnH);

  // out-projection: attn_output = attnH @ Wo^T + bo  (fp32 out)
  gemm_outproj_kernel<<<dim3(EMB / 256, MROWS / 32), 256, 0, stream>>>(
      attnH, wobf, bo, outAttn);
}
